// MultiHeadAttentionBlock_50903952392799
// MI455X (gfx1250) — matrix operand order repaired and checked by reading, not compiled, not run on hardware
//
#include <hip/hip_runtime.h>

#define DM   1024
#define KDIM 1024
#define SEQ  2048
#define NH   16
#define DK   64

typedef __attribute__((ext_vector_type(16))) __bf16 v16bf;
typedef __attribute__((ext_vector_type(8)))  __bf16 v8bf;
typedef __attribute__((ext_vector_type(8)))  float  v8f;

#if defined(__HIP_DEVICE_COMPILE__) && __has_builtin(__builtin_amdgcn_global_load_async_to_lds_b128)
#define HAVE_ASYNC 1
typedef __attribute__((ext_vector_type(4))) int i32x4;
typedef __attribute__((address_space(1))) i32x4 gi32x4;   // global int4
typedef __attribute__((address_space(3))) i32x4 li32x4;   // LDS int4
#else
#define HAVE_ASYNC 0
#endif

__device__ __forceinline__ v8f vzero8() {
    v8f z;
#pragma unroll
    for (int i = 0; i < 8; ++i) z[i] = 0.0f;
    return z;
}

// 16-byte global -> LDS copy: async (gfx1250 GLOBAL_LOAD_ASYNC_TO_LDS_B128,
// tracked by ASYNCcnt) when available, else plain load + ds_store.
__device__ __forceinline__ void copy16_g2l(const void* g, void* l) {
#if HAVE_ASYNC
    __builtin_amdgcn_global_load_async_to_lds_b128(
        (gi32x4*)(void*)g, (li32x4*)l, 0, 0);
#else
    *(uint4*)l = *(const uint4*)g;
#endif
}

__device__ __forceinline__ void async_wait0() {
#if HAVE_ASYNC
#if __has_builtin(__builtin_amdgcn_s_wait_asynccnt)
    __builtin_amdgcn_s_wait_asynccnt(0);
#else
    asm volatile("s_wait_asynccnt 0x0" ::: "memory");
#endif
#endif
}

// ---------------------------------------------------------------------------
// Tiled GEMM: C[8192,1024] = A[8192,1024] @ W[1024,1024]^T + bias
// mode 0: write bf16 [B,H,S,DK]     (Q / K projection)
// mode 2: write bf16 [B,H,DK,S]     (V projection, pre-transposed for P·V)
// mode 3: write f32  [8192,1024]    (final output projection)
// Block: 256 thr = 8 waves, block tile 128x128, wave tile 32x64, K-step 32.
// Double-buffered LDS: next K-slice prefetched into registers during WMMA.
// ---------------------------------------------------------------------------
__global__ __launch_bounds__(256)
void proj_gemm(const float* __restrict__ A, const float* __restrict__ W,
               const float* __restrict__ bias, __bf16* __restrict__ outb,
               float* __restrict__ outf, int mode)
{
    __shared__ __bf16 As[2][128][48];   // row stride 48 bf16 = 96 B
    __shared__ __bf16 Bs[2][128][48];

    const int t     = threadIdx.x;
    const int lane  = t & 31;
    const int wv    = t >> 5;        // 0..7
    const int waveM = wv >> 1;       // 0..3
    const int waveN = wv & 1;        // 0..1
    const int lm    = lane & 15;
    const int lh    = lane >> 4;     // 0/1

    const int mBase = blockIdx.x * 128;
    const int nBase = blockIdx.y * 128;

    v8f acc[2][4];
#pragma unroll
    for (int i = 0; i < 2; ++i)
#pragma unroll
        for (int j = 0; j < 4; ++j) acc[i][j] = vzero8();

    float4 ra[4], rw[4];
    int rowv[4], c4v[4];
#pragma unroll
    for (int it = 0; it < 4; ++it) {
        int fi   = t + it * 256;             // 0..1023 float4 slots
        rowv[it] = fi >> 3;
        c4v[it]  = (fi & 7) * 4;
    }

    // prologue: load + stage K-slice 0 into buffer 0
#pragma unroll
    for (int it = 0; it < 4; ++it) {
        ra[it] = *(const float4*)(A + (size_t)(mBase + rowv[it]) * KDIM + c4v[it]);
        rw[it] = *(const float4*)(W + (size_t)(nBase + rowv[it]) * KDIM + c4v[it]);
    }
#pragma unroll
    for (int it = 0; it < 4; ++it) {
        __bf16* da = &As[0][rowv[it]][c4v[it]];
        da[0] = (__bf16)ra[it].x; da[1] = (__bf16)ra[it].y;
        da[2] = (__bf16)ra[it].z; da[3] = (__bf16)ra[it].w;
        __bf16* dw = &Bs[0][rowv[it]][c4v[it]];
        dw[0] = (__bf16)rw[it].x; dw[1] = (__bf16)rw[it].y;
        dw[2] = (__bf16)rw[it].z; dw[3] = (__bf16)rw[it].w;
    }

    int cur = 0;
    for (int k0 = 0; k0 < KDIM; k0 += 32) {
        __syncthreads();
        const bool hasNext = (k0 + 32) < KDIM;
        if (hasNext) {       // prefetch next slice into registers (overlaps WMMA)
#pragma unroll
            for (int it = 0; it < 4; ++it) {
                ra[it] = *(const float4*)(A + (size_t)(mBase + rowv[it]) * KDIM + k0 + 32 + c4v[it]);
                rw[it] = *(const float4*)(W + (size_t)(nBase + rowv[it]) * KDIM + k0 + 32 + c4v[it]);
            }
        }

        // B fragments: lane holds col n=lm, K = lh*16 + 0..15 (contiguous 32B)
        v16bf bfrag[4];
#pragma unroll
        for (int j = 0; j < 4; ++j)
            bfrag[j] = *(const v16bf*)&Bs[cur][waveN * 64 + j * 16 + lm][lh * 16];

        // A fragments: lane holds row m=lm, K = lh*16 + 0..15 (contiguous)
        v16bf afrag[2];
#pragma unroll
        for (int i = 0; i < 2; ++i) {
            const __bf16* pa = &As[cur][waveM * 32 + i * 16 + lm][0];
            union { v16bf v; v8bf h[2]; } u;
            u.h[0] = *(const v8bf*)(pa + lh * 16);
            u.h[1] = *(const v8bf*)(pa + lh * 16 + 8);
            afrag[i] = u.v;
        }

#pragma unroll
        for (int i = 0; i < 2; ++i)
#pragma unroll
            for (int j = 0; j < 4; ++j)
                acc[i][j] = __builtin_amdgcn_wmma_f32_16x16x32_bf16(
                    false, afrag[i], false, bfrag[j], (short)0, acc[i][j], false, false);

        if (hasNext) {       // stage prefetched slice into the other buffer
            const int nb = cur ^ 1;
#pragma unroll
            for (int it = 0; it < 4; ++it) {
                __bf16* da = &As[nb][rowv[it]][c4v[it]];
                da[0] = (__bf16)ra[it].x; da[1] = (__bf16)ra[it].y;
                da[2] = (__bf16)ra[it].z; da[3] = (__bf16)ra[it].w;
                __bf16* dw = &Bs[nb][rowv[it]][c4v[it]];
                dw[0] = (__bf16)rw[it].x; dw[1] = (__bf16)rw[it].y;
                dw[2] = (__bf16)rw[it].z; dw[3] = (__bf16)rw[it].w;
            }
        }
        cur ^= 1;
    }

    float bn[4];
#pragma unroll
    for (int j = 0; j < 4; ++j) bn[j] = bias[nBase + waveN * 64 + j * 16 + lm];

#pragma unroll
    for (int i = 0; i < 2; ++i) {
#pragma unroll
        for (int j = 0; j < 4; ++j) {
            int n = nBase + waveN * 64 + j * 16 + lm;
#pragma unroll
            for (int r = 0; r < 8; ++r) {
                int m = mBase + waveM * 32 + i * 16 + r + 8 * lh;
                float c = acc[i][j][r] + bn[j];
                if (mode == 3) {
                    outf[(size_t)m * DM + n] = c;
                } else {
                    int b = m >> 11, s = m & 2047;
                    int h = n >> 6,  d = n & 63;
                    size_t idx;
                    if (mode == 2)
                        idx = ((size_t)(b * NH + h) * DK + d) * SEQ + s;   // [B,H,DK,S]
                    else
                        idx = ((size_t)(b * NH + h) * SEQ + s) * DK + d;   // [B,H,S,DK]
                    outb[idx] = (__bf16)c;
                }
            }
        }
    }
}

// ---------------------------------------------------------------------------
// Causal flash attention: grid (S/128, B*H), 256 thr = 8 waves.
// Wave w owns queries qBase + 16w .. +15 -> softmax is wave-local.
// K/V^T tiles double-buffered; staged via async global->LDS copies (ASYNCcnt).
// Schedule per tile: s_wait_asynccnt 0 -> barrier -> issue(next) -> compute(cur).
// ---------------------------------------------------------------------------
__global__ __launch_bounds__(256)
void attn_kernel(const __bf16* __restrict__ Q, const __bf16* __restrict__ K,
                 const __bf16* __restrict__ VT, float* __restrict__ O)
{
    __shared__ __bf16 Ks[2][64][64];     // [buf][key][d]
    __shared__ __bf16 VTs[2][64][64];    // [buf][d][key]
    __shared__ __bf16 Pbuf[8][16][64];   // per-wave P re-layout buffer

    const int t    = threadIdx.x;
    const int lane = t & 31;
    const int wv   = t >> 5;
    const int lm   = lane & 15;
    const int lh   = lane >> 4;

    const int bh    = blockIdx.y;
    const int b     = bh >> 4, h = bh & 15;
    const int qBase = blockIdx.x * 128;

    const __bf16* Qh = Q  + (size_t)bh * SEQ * DK;
    const __bf16* Kh = K  + (size_t)bh * SEQ * DK;
    const __bf16* Vh = VT + (size_t)bh * DK * SEQ;

    // Resident Q fragments (16 queries x 64 d = two 16x32 A-frags)
    v16bf qa[2];
    {
        const __bf16* qrow = Qh + (size_t)(qBase + wv * 16 + lm) * DK;
        union { v16bf v; v8bf hh[2]; } u;
        u.hh[0] = *(const v8bf*)(qrow + lh * 16);
        u.hh[1] = *(const v8bf*)(qrow + lh * 16 + 8);
        qa[0] = u.v;
        u.hh[0] = *(const v8bf*)(qrow + 32 + lh * 16);
        u.hh[1] = *(const v8bf*)(qrow + 32 + lh * 16 + 8);
        qa[1] = u.v;
    }

    float mi[8], li[8];
    v8f o[4];
#pragma unroll
    for (int r = 0; r < 8; ++r) { mi[r] = -1e30f; li[r] = 0.0f; }
#pragma unroll
    for (int j = 0; j < 4; ++j) o[j] = vzero8();

    // Stage one 64-key tile (K rows contiguous: 8 KB; VT rows strided 128 B)
    auto stage = [&](int kt, int buf) {
        const char* srcK = (const char*)(Kh + (size_t)kt * DK);
        char* dstK = (char*)&Ks[buf][0][0];
        copy16_g2l(srcK + t * 16,          dstK + t * 16);
        copy16_g2l(srcK + (t + 256) * 16,  dstK + (t + 256) * 16);
#pragma unroll
        for (int it = 0; it < 2; ++it) {
            int idx = t + it * 256;
            int dr  = idx >> 3;
            int c   = idx & 7;
            copy16_g2l(Vh + (size_t)dr * SEQ + kt + c * 8, &VTs[buf][dr][c * 8]);
        }
    };

    const int kEnd = qBase + 128;
    stage(0, 0);
    int cur = 0;

    for (int kt = 0; kt < kEnd; kt += 64) {
        async_wait0();        // own async writes to LDS complete
        __syncthreads();      // everyone's writes visible, prior reads done
        if (kt + 64 < kEnd) stage(kt + 64, cur ^ 1);   // overlap with compute

        // S = Q K^T for 64 keys (4 subtiles x 2 WMMAs over d)
        v8f sf[4];
#pragma unroll
        for (int kk = 0; kk < 4; ++kk) {
            v8f s = vzero8();
            v16bf kb0 = *(const v16bf*)&Ks[cur][kk * 16 + lm][lh * 16];
            v16bf kb1 = *(const v16bf*)&Ks[cur][kk * 16 + lm][32 + lh * 16];
            s = __builtin_amdgcn_wmma_f32_16x16x32_bf16(false, qa[0], false, kb0, (short)0, s, false, false);
            s = __builtin_amdgcn_wmma_f32_16x16x32_bf16(false, qa[1], false, kb1, (short)0, s, false, false);
            sf[kk] = s;
        }

        // Scale + causal mask
#pragma unroll
        for (int kk = 0; kk < 4; ++kk) {
            int key = kt + kk * 16 + lm;
#pragma unroll
            for (int r = 0; r < 8; ++r) {
                int qi = qBase + wv * 16 + r + 8 * lh;
                float sv = sf[kk][r] * 0.125f;        // 1/sqrt(64)
                if (key > qi) sv = -1e30f;
                sf[kk][r] = sv;
            }
        }

        // Online softmax per query row (rows live in 16-lane halves)
#pragma unroll
        for (int r = 0; r < 8; ++r) {
            float rm = fmaxf(fmaxf(sf[0][r], sf[1][r]), fmaxf(sf[2][r], sf[3][r]));
#pragma unroll
            for (int mk = 1; mk < 16; mk <<= 1)
                rm = fmaxf(rm, __shfl_xor(rm, mk, 32));
            float mnew  = fmaxf(mi[r], rm);
            float alpha = __expf(mi[r] - mnew);
            mi[r] = mnew;
            float rs = 0.0f;
#pragma unroll
            for (int kk = 0; kk < 4; ++kk) {
                float sv = sf[kk][r];
                float p  = (sv > -1e29f) ? __expf(sv - mnew) : 0.0f;  // guard fully-masked rows
                rs += p;
                Pbuf[wv][r + 8 * lh][kk * 16 + lm] = (__bf16)p;
            }
#pragma unroll
            for (int mk = 1; mk < 16; mk <<= 1)
                rs += __shfl_xor(rs, mk, 32);
            li[r] = li[r] * alpha + rs;
#pragma unroll
            for (int j = 0; j < 4; ++j) o[j][r] *= alpha;
        }

        // P·V: reload P as A-fragments (same-wave LDS is in-order, no barrier)
        v16bf pa0, pa1;
        {
            const __bf16* prow = &Pbuf[wv][lm][0];
            union { v16bf v; v8bf hh[2]; } u;
            u.hh[0] = *(const v8bf*)(prow + lh * 16);
            u.hh[1] = *(const v8bf*)(prow + lh * 16 + 8);
            pa0 = u.v;
            u.hh[0] = *(const v8bf*)(prow + 32 + lh * 16);
            u.hh[1] = *(const v8bf*)(prow + 32 + lh * 16 + 8);
            pa1 = u.v;
        }
#pragma unroll
        for (int j = 0; j < 4; ++j) {
            v16bf vb0 = *(const v16bf*)&VTs[cur][j * 16 + lm][lh * 16];
            v16bf vb1 = *(const v16bf*)&VTs[cur][j * 16 + lm][32 + lh * 16];
            o[j] = __builtin_amdgcn_wmma_f32_16x16x32_bf16(false, pa0, false, vb0, (short)0, o[j], false, false);
            o[j] = __builtin_amdgcn_wmma_f32_16x16x32_bf16(false, pa1, false, vb1, (short)0, o[j], false, false);
        }
        cur ^= 1;
    }

    // Normalize and write merged-head f32 output [B*S, H*DK]
#pragma unroll
    for (int r = 0; r < 8; ++r) {
        float inv = 1.0f / li[r];
        int qi = qBase + wv * 16 + r + 8 * lh;
        size_t base = (size_t)(b * SEQ + qi) * DM + h * DK;
#pragma unroll
        for (int j = 0; j < 4; ++j)
            O[base + j * 16 + lm] = o[j][r] * inv;
    }
}

// ---------------------------------------------------------------------------
extern "C" void kernel_launch(void* const* d_in, const int* in_sizes, int n_in,
                              void* d_out, int out_size, void* d_ws, size_t ws_size,
                              hipStream_t stream)
{
    const float* q  = (const float*)d_in[0];
    const float* k  = (const float*)d_in[1];
    const float* v  = (const float*)d_in[2];
    // d_in[3] = causal mask (tril) -> causality is hardcoded in attn_kernel
    const float* wq = (const float*)d_in[4];
    const float* bq = (const float*)d_in[5];
    const float* wk = (const float*)d_in[6];
    const float* bk = (const float*)d_in[7];
    const float* wv = (const float*)d_in[8];
    const float* bv = (const float*)d_in[9];
    const float* wo = (const float*)d_in[10];
    const float* bo = (const float*)d_in[11];
    float* out = (float*)d_out;

    char* ws = (char*)d_ws;
    __bf16* Qbf  = (__bf16*)(ws);                        // 16 MB
    __bf16* Kbf  = (__bf16*)(ws + (size_t)16 * 1048576); // 16 MB
    __bf16* VTbf = (__bf16*)(ws + (size_t)32 * 1048576); // 16 MB
    float*  Of   = (float*) (ws + (size_t)48 * 1048576); // 32 MB

    dim3 gridG(64, 8), blockG(256);
    proj_gemm<<<gridG, blockG, 0, stream>>>(q, wq, bq, Qbf,  nullptr, 0);
    proj_gemm<<<gridG, blockG, 0, stream>>>(k, wk, bk, Kbf,  nullptr, 0);
    proj_gemm<<<gridG, blockG, 0, stream>>>(v, wv, bv, VTbf, nullptr, 2);

    attn_kernel<<<dim3(16, 64), dim3(256), 0, stream>>>(Qbf, Kbf, VTbf, Of);

    proj_gemm<<<gridG, blockG, 0, stream>>>(Of, wo, bo, nullptr, out, 3);
}
